// MultiHeadAttention_17102559773005
// MI455X (gfx1250) — matrix operand order repaired and checked by reading, not compiled, not run on hardware
//
#include <hip/hip_runtime.h>
#include <hip/hip_fp16.h>

// ---------------------------------------------------------------------------
// MHA forward for MI455X (gfx1250): f16 WMMA pipeline, flash-attention core.
//   B=2, S=2048, H=1024, heads=16, head_dim=64, causal mask.
// Stages: cvt(f32->f16) -> QKV gemm -> flash attn -> out-proj gemm.
// GEMMs use double-buffered LDS fed by async global->LDS DMA when available.
// ---------------------------------------------------------------------------

typedef __attribute__((ext_vector_type(16))) _Float16 v16h;
typedef __attribute__((ext_vector_type(8)))  _Float16 h8;
typedef __attribute__((ext_vector_type(4)))  _Float16 h4;
typedef __attribute__((ext_vector_type(8)))  float    v8f;
typedef __attribute__((ext_vector_type(4)))  int      i4;

#define BATCH 2
#define SEQ   2048
#define HID   1024
#define NH    16
#define HD    64
#define MROWS (BATCH * SEQ)   // 4096

// ---- CDNA5 async global->LDS copy (ASYNCcnt) with safe fallback -----------
#if defined(__has_builtin)
#if __has_builtin(__builtin_amdgcn_global_load_async_to_lds_b128)
#define HAVE_ASYNC_LDS 1
#endif
#endif
#ifndef HAVE_ASYNC_LDS
#define HAVE_ASYNC_LDS 0
#endif

typedef __attribute__((address_space(1))) i4* gptr_i4;
typedef __attribute__((address_space(3))) i4* lptr_i4;

__device__ __forceinline__ void async_copy16(const _Float16* g, _Float16* l) {
#if HAVE_ASYNC_LDS
    __builtin_amdgcn_global_load_async_to_lds_b128(
        (gptr_i4)(i4*)(void*)const_cast<_Float16*>(g),
        (lptr_i4)(i4*)(void*)l,
        0, 0);
#else
    *(h8*)l = *(const h8*)g;   // sync fallback: global_load_b128 + ds_store_b128
#endif
}

#if defined(__has_builtin) && __has_builtin(__builtin_amdgcn_s_wait_asynccnt)
#define WAIT_ASYNC(n) __builtin_amdgcn_s_wait_asynccnt(n)
#else
#define WAIT_ASYNC(n) asm volatile("s_wait_asynccnt %0" ::"i"(n) : "memory")
#endif

__device__ __forceinline__ v8f v8f_zero() {
    v8f z;
#pragma unroll
    for (int e = 0; e < 8; ++e) z[e] = 0.0f;
    return z;
}

__device__ __forceinline__ v16h make_a_frag(const _Float16* p /*row base + kbase*/) {
    // A 16x32 f16 layout: lane row = lane&15, kbase = (lane>>4)*8 (+ caller's kc*32).
    // elements 0..7 = K kbase..kbase+7 ; elements 8..15 = K kbase+16..kbase+23.
    h8 lo = *(const h8*)p;
    h8 hh = *(const h8*)(p + 16);
    v16h a;
#pragma unroll
    for (int e = 0; e < 8; ++e) { a[e] = lo[e]; a[e + 8] = hh[e]; }
    return a;
}

__device__ __forceinline__ v16h make_b_frag(const _Float16* p /*n-row base + (lane>>4)*8*/) {
    // B 32x16 f16 layout: col = lane&15, elements 0..7 = K (lane>>4)*8+e ; 8..15 = K (lane>>4)*8+16+e.
    h8 lo = *(const h8*)p;
    h8 hh = *(const h8*)(p + 16);
    v16h b;
#pragma unroll
    for (int e = 0; e < 8; ++e) { b[e] = lo[e]; b[e + 8] = hh[e]; }
    return b;
}

__device__ __forceinline__ v8f wmma_f16(v16h a, v16h b, v8f c) {
    return __builtin_amdgcn_wmma_f32_16x16x32_f16(false, a, false, b, (short)0, c, false, false);
}

// ---------------------------------------------------------------------------
// Stage 1: f32 -> f16 conversion (n divisible by 1024)
// ---------------------------------------------------------------------------
__global__ __launch_bounds__(256) void cvt_f32_to_f16(const float* __restrict__ src,
                                                      _Float16* __restrict__ dst, int n4) {
    int i = blockIdx.x * 256 + threadIdx.x;
    if (i < n4) {
        float4 v = ((const float4*)src)[i];
        h4 o;
        o[0] = (_Float16)v.x; o[1] = (_Float16)v.y;
        o[2] = (_Float16)v.z; o[3] = (_Float16)v.w;
        ((h4*)dst)[i] = o;
    }
}

// ---------------------------------------------------------------------------
// Shared GEMM mainloop: C(128x128 tile) = A[M x 1024] * W[N x 1024]^T
// 4 waves, each a 64x64 sub-tile with 16 WMMA accumulators.
// Double-buffered LDS; async DMA of slab i+1 overlaps WMMA of slab i.
// ---------------------------------------------------------------------------
__device__ __forceinline__ void gemm_mainloop(const _Float16* __restrict__ A,
                                              const _Float16* __restrict__ W,
                                              _Float16 (*lA)[128][32],
                                              _Float16 (*lB)[128][32],
                                              int tileM, int tileN,
                                              int wm, int wn, int tid, int l16, int hi,
                                              v8f acc[4][4]) {
    auto issue = [&](int kk, int buf) {
#pragma unroll
        for (int p4 = 0; p4 < 4; ++p4) {
            int idx = p4 * 128 + tid;
            int r = idx >> 2, c = (idx & 3) * 8;
            async_copy16(&A[(size_t)(tileM + r) * HID + kk + c], &lA[buf][r][c]);
            async_copy16(&W[(size_t)(tileN + r) * HID + kk + c], &lB[buf][r][c]);
        }
    };

    issue(0, 0);
    for (int it = 0; it < HID / 32; ++it) {
        const int buf = it & 1;
        if (it + 1 < HID / 32) {
            issue((it + 1) * 32, buf ^ 1);   // DMA next slab into other buffer
            WAIT_ASYNC(8);                   // our previous 8 async ops have landed
        } else {
            WAIT_ASYNC(0);
        }
        __syncthreads();                     // everyone's slab `it` is in LDS

        v16h af[4], bf[4];
#pragma unroll
        for (int mi = 0; mi < 4; ++mi)
            af[mi] = make_a_frag(&lA[buf][wm + mi * 16 + l16][hi * 8]);
#pragma unroll
        for (int ni = 0; ni < 4; ++ni)
            bf[ni] = make_b_frag(&lB[buf][wn + ni * 16 + l16][hi * 8]);
#pragma unroll
        for (int mi = 0; mi < 4; ++mi)
#pragma unroll
            for (int ni = 0; ni < 4; ++ni)
                acc[mi][ni] = wmma_f16(af[mi], bf[ni], acc[mi][ni]);
        __syncthreads();                     // reads done before buf is re-DMA'd
    }
}

// ---------------------------------------------------------------------------
// Stage 2: QKV projection. Out columns 0..1023 -> Q, 1024..2047 -> K, 2048.. -> V,
// scattered into [B][head][S][hd] f16 layout.
// ---------------------------------------------------------------------------
__global__ __launch_bounds__(128) void gemm_qkv_kernel(const _Float16* __restrict__ A,
                                                       const _Float16* __restrict__ W,
                                                       _Float16* __restrict__ Qo,
                                                       _Float16* __restrict__ Ko,
                                                       _Float16* __restrict__ Vo) {
    __shared__ _Float16 lA[2][128][32];
    __shared__ _Float16 lB[2][128][32];
    const int tid = threadIdx.x, wid = tid >> 5, lane = tid & 31;
    const int l16 = lane & 15, hi = lane >> 4;
    const int tileN = blockIdx.x * 128, tileM = blockIdx.y * 128;
    const int wm = (wid >> 1) * 64, wn = (wid & 1) * 64;

    v8f acc[4][4];
#pragma unroll
    for (int mi = 0; mi < 4; ++mi)
#pragma unroll
        for (int ni = 0; ni < 4; ++ni) acc[mi][ni] = v8f_zero();

    gemm_mainloop(A, W, lA, lB, tileM, tileN, wm, wn, tid, l16, hi, acc);

#pragma unroll
    for (int mi = 0; mi < 4; ++mi)
#pragma unroll
        for (int ni = 0; ni < 4; ++ni)
#pragma unroll
            for (int r = 0; r < 8; ++r) {
                int m = tileM + wm + mi * 16 + hi * 8 + r;
                int n = tileN + wn + ni * 16 + l16;
                int which = n >> 10;
                int hn = n & 1023;
                int head = hn >> 6, d = hn & 63;
                int bb = m >> 11, s = m & 2047;
                _Float16* dst = (which == 0) ? Qo : (which == 1) ? Ko : Vo;
                dst[(((size_t)bb * NH + head) * SEQ + s) * HD + d] = (_Float16)acc[mi][ni][r];
            }
}

// ---------------------------------------------------------------------------
// Stage 3: flash attention. grid = (qblocks=16, heads=16, batch=2), 128 threads.
// 4 waves share LDS K/V tiles; each wave owns a 32-row q tile, online softmax.
// ---------------------------------------------------------------------------
__global__ __launch_bounds__(128) void attn_kernel(const _Float16* __restrict__ Qg,
                                                   const _Float16* __restrict__ Kg,
                                                   const _Float16* __restrict__ Vg,
                                                   _Float16* __restrict__ Og) {
    __shared__ _Float16 lK[64][64];      // [key][d]   -> B frags for Q*K^T
    __shared__ _Float16 lV[64][64];      // [d][key]   -> B frags for P*V
    __shared__ _Float16 lP[4][32][64];   // per-wave P tile (A layout source)

    const int tid = threadIdx.x, wid = tid >> 5, lane = tid & 31;
    const int l16 = lane & 15, hi = lane >> 4;
    const int qblk = blockIdx.x, head = blockIdx.y, bat = blockIdx.z;
    const size_t bh = ((size_t)(bat * NH + head)) * SEQ * HD;
    const _Float16* Qb = Qg + bh;
    const _Float16* Kb = Kg + bh;
    const _Float16* Vb = Vg + bh;
    const int qt = qblk * 128 + wid * 32;

    // Q fragments held in registers for the whole kernel: [mi][kc]
    v16h aq[2][2];
#pragma unroll
    for (int mi = 0; mi < 2; ++mi)
#pragma unroll
        for (int kc = 0; kc < 2; ++kc)
            aq[mi][kc] = make_a_frag(&Qb[(size_t)(qt + mi * 16 + l16) * HD + kc * 32 + hi * 8]);

    float mrow[2][8], lrow[2][8];
    v8f oacc[2][4];
#pragma unroll
    for (int mi = 0; mi < 2; ++mi) {
#pragma unroll
        for (int r = 0; r < 8; ++r) { mrow[mi][r] = -1e30f; lrow[mi][r] = 0.0f; }
#pragma unroll
        for (int dn = 0; dn < 4; ++dn) oacc[mi][dn] = v8f_zero();
    }

    const int nkt = qblk * 2 + 2;   // causal: keys in [0, qblk*128+128)
    for (int kt = 0; kt < nkt; ++kt) {
        const int ks = kt * 64;
        // cooperative load of K tile (row major) and V tile (transposed [d][key])
#pragma unroll
        for (int p4 = 0; p4 < 4; ++p4) {
            int idx = p4 * 128 + tid;
            int r = idx >> 3, c = (idx & 7) * 8;
            *(h8*)&lK[r][c] = *(const h8*)&Kb[(size_t)(ks + r) * HD + c];
            h8 tv = *(const h8*)&Vb[(size_t)(ks + r) * HD + c];
#pragma unroll
            for (int j = 0; j < 8; ++j) lV[c + j][r] = tv[j];
        }
        __syncthreads();

        if (ks <= qt + 31) {   // wave-uniform: EXEC stays all-ones for WMMA
            // ---- scores S = Q * K^T over this 64-key tile ----
            v8f sacc[2][4];
#pragma unroll
            for (int mi = 0; mi < 2; ++mi)
#pragma unroll
                for (int ni = 0; ni < 4; ++ni) sacc[mi][ni] = v8f_zero();
#pragma unroll
            for (int kc = 0; kc < 2; ++kc)
#pragma unroll
                for (int ni = 0; ni < 4; ++ni) {
                    v16h bk = make_b_frag(&lK[ni * 16 + l16][kc * 32 + hi * 8]);
#pragma unroll
                    for (int mi = 0; mi < 2; ++mi)
                        sacc[mi][ni] = wmma_f16(aq[mi][kc], bk, sacc[mi][ni]);
                }

            // ---- online softmax update + emit P (f16) into per-wave LDS ----
#pragma unroll
            for (int mi = 0; mi < 2; ++mi)
#pragma unroll
                for (int r = 0; r < 8; ++r) {
                    const int mg = qt + mi * 16 + hi * 8 + r;
                    float pv[4];
                    float rm = -1e30f;
#pragma unroll
                    for (int ni = 0; ni < 4; ++ni) {
                        float v = sacc[mi][ni][r] * 0.125f;   // 1/sqrt(64)
                        int kg = ks + ni * 16 + l16;
                        if (kg > mg) v = -1e30f;              // causal mask
                        pv[ni] = v;
                        rm = fmaxf(rm, v);
                    }
#pragma unroll
                    for (int off = 8; off >= 1; off >>= 1)
                        rm = fmaxf(rm, __shfl_xor(rm, off, 32));   // within 16-lane half
                    float nm = fmaxf(mrow[mi][r], rm);
                    float corr = __expf(mrow[mi][r] - nm);
                    mrow[mi][r] = nm;
                    float rs = 0.0f;
#pragma unroll
                    for (int ni = 0; ni < 4; ++ni) {
                        float e = __expf(pv[ni] - nm);
                        pv[ni] = e;
                        rs += e;
                    }
#pragma unroll
                    for (int off = 8; off >= 1; off >>= 1)
                        rs += __shfl_xor(rs, off, 32);
                    lrow[mi][r] = lrow[mi][r] * corr + rs;
#pragma unroll
                    for (int dn = 0; dn < 4; ++dn) oacc[mi][dn][r] *= corr;
#pragma unroll
                    for (int ni = 0; ni < 4; ++ni)
                        lP[wid][mi * 16 + hi * 8 + r][ni * 16 + l16] = (_Float16)pv[ni];
                }
            // cross-lane LDS RAW inside the same wave: drain DS before re-reading P
            asm volatile("s_wait_dscnt 0" ::: "memory");

            // ---- O += P * V ----
#pragma unroll
            for (int kc = 0; kc < 2; ++kc) {
                v16h ap[2];
#pragma unroll
                for (int mi = 0; mi < 2; ++mi)
                    ap[mi] = make_a_frag(&lP[wid][mi * 16 + l16][kc * 32 + hi * 8]);
#pragma unroll
                for (int dn = 0; dn < 4; ++dn) {
                    v16h bv = make_b_frag(&lV[dn * 16 + l16][kc * 32 + hi * 8]);
#pragma unroll
                    for (int mi = 0; mi < 2; ++mi)
                        oacc[mi][dn] = wmma_f16(ap[mi], bv, oacc[mi][dn]);
                }
            }
        }
        __syncthreads();
    }

    // ---- normalize and write O as f16 in [B*S][H] layout (head-merged) ----
#pragma unroll
    for (int mi = 0; mi < 2; ++mi)
#pragma unroll
        for (int r = 0; r < 8; ++r) {
            float inv = 1.0f / lrow[mi][r];
            int s = qt + mi * 16 + hi * 8 + r;
            size_t rowoff = ((size_t)(bat * SEQ + s)) * HID + head * HD;
#pragma unroll
            for (int dn = 0; dn < 4; ++dn)
                Og[rowoff + dn * 16 + l16] = (_Float16)(oacc[mi][dn][r] * inv);
        }
}

// ---------------------------------------------------------------------------
// Stage 4: output projection, writes f32 directly to d_out.
// ---------------------------------------------------------------------------
__global__ __launch_bounds__(128) void gemm_out_kernel(const _Float16* __restrict__ A,
                                                       const _Float16* __restrict__ W,
                                                       float* __restrict__ out) {
    __shared__ _Float16 lA[2][128][32];
    __shared__ _Float16 lB[2][128][32];
    const int tid = threadIdx.x, wid = tid >> 5, lane = tid & 31;
    const int l16 = lane & 15, hi = lane >> 4;
    const int tileN = blockIdx.x * 128, tileM = blockIdx.y * 128;
    const int wm = (wid >> 1) * 64, wn = (wid & 1) * 64;

    v8f acc[4][4];
#pragma unroll
    for (int mi = 0; mi < 4; ++mi)
#pragma unroll
        for (int ni = 0; ni < 4; ++ni) acc[mi][ni] = v8f_zero();

    gemm_mainloop(A, W, lA, lB, tileM, tileN, wm, wn, tid, l16, hi, acc);

#pragma unroll
    for (int mi = 0; mi < 4; ++mi)
#pragma unroll
        for (int ni = 0; ni < 4; ++ni)
#pragma unroll
            for (int r = 0; r < 8; ++r) {
                int m = tileM + wm + mi * 16 + hi * 8 + r;
                int n = tileN + wn + ni * 16 + l16;
                out[(size_t)m * HID + n] = acc[mi][ni][r];
            }
}

// ---------------------------------------------------------------------------
// Host launch
// ---------------------------------------------------------------------------
extern "C" void kernel_launch(void* const* d_in, const int* in_sizes, int n_in,
                              void* d_out, int out_size, void* d_ws, size_t ws_size,
                              hipStream_t stream) {
    const float* x     = (const float*)d_in[0];
    // d_in[1] = attention_mask (causal, known statically) -- unused
    const float* w_qkv = (const float*)d_in[2];
    const float* w_out = (const float*)d_in[3];
    float* out = (float*)d_out;

    _Float16* ws = (_Float16*)d_ws;
    size_t off = 0;
    _Float16* xh    = ws + off; off += (size_t)MROWS * HID;       // 4M halfs
    _Float16* wqkvh = ws + off; off += (size_t)3 * HID * HID;     // 3M halfs
    _Float16* wouth = ws + off; off += (size_t)HID * HID;         // 1M halfs
    _Float16* qh    = ws + off; off += (size_t)MROWS * HID;
    _Float16* kh    = ws + off; off += (size_t)MROWS * HID;
    _Float16* vh    = ws + off; off += (size_t)MROWS * HID;
    _Float16* oh    = ws + off; off += (size_t)MROWS * HID;       // total ~50 MB

    {   // conversions (all sizes divisible by 1024)
        int n;
        n = MROWS * HID;       cvt_f32_to_f16<<<n / 4 / 256, 256, 0, stream>>>(x, xh, n / 4);
        n = 3 * HID * HID;     cvt_f32_to_f16<<<n / 4 / 256, 256, 0, stream>>>(w_qkv, wqkvh, n / 4);
        n = HID * HID;         cvt_f32_to_f16<<<n / 4 / 256, 256, 0, stream>>>(w_out, wouth, n / 4);
    }

    // QKV projection: M=4096, N=3072 -> grid (24, 32)
    gemm_qkv_kernel<<<dim3(3 * HID / 128, MROWS / 128), 128, 0, stream>>>(xh, wqkvh, qh, kh, vh);

    // Flash attention: grid (qblocks=16, heads=16, batch=2)
    attn_kernel<<<dim3(SEQ / 128, NH, BATCH), 128, 0, stream>>>(qh, kh, vh, oh);

    // Output projection: M=4096, N=1024 -> grid (8, 32)
    gemm_out_kernel<<<dim3(HID / 128, MROWS / 128), 128, 0, stream>>>(oh, wouth, out);
}
